// MambaBlockRef2_31971736552196
// MI455X (gfx1250) — matrix operand order repaired and checked by reading, not compiled, not run on hardware
//
#include <hip/hip_runtime.h>
#include <hip/hip_bf16.h>

typedef __attribute__((ext_vector_type(16))) __bf16 v16bf;
typedef __attribute__((ext_vector_type(8)))  __bf16 v8bf;
typedef __attribute__((ext_vector_type(8)))  float  v8f;

#define D_MODEL 1024
#define D_INNER 2048
#define D_STATE 16
#define D_CONV  4
#define DT_RANK 64
#define B_SZ    2
#define L_SEQ   2048
#define M_ROWS  (B_SZ * L_SEQ)           // 4096
#define XDBL_N  (DT_RANK + 2 * D_STATE)  // 96

// ---------------------------------------------------------------------------
// LayerNorm: one block per row of 1024, eps=1e-6, biased variance
// ---------------------------------------------------------------------------
__global__ __launch_bounds__(256) void ln_kernel(const float* __restrict__ x,
                                                 const float* __restrict__ gamma,
                                                 const float* __restrict__ beta,
                                                 float* __restrict__ h) {
  int m = blockIdx.x;
  int t = threadIdx.x;
  const float* row = x + (size_t)m * D_MODEL;
  float v[4];
  float s = 0.f, ss = 0.f;
#pragma unroll
  for (int i = 0; i < 4; ++i) {
    v[i] = row[t + 256 * i];
    s += v[i];
    ss += v[i] * v[i];
  }
  __shared__ float red0[256];
  __shared__ float red1[256];
  red0[t] = s; red1[t] = ss;
  __syncthreads();
  for (int off = 128; off > 0; off >>= 1) {
    if (t < off) { red0[t] += red0[t + off]; red1[t] += red1[t + off]; }
    __syncthreads();
  }
  float mu  = red0[0] * (1.0f / D_MODEL);
  float var = red1[0] * (1.0f / D_MODEL) - mu * mu;
  float rstd = rsqrtf(var + 1e-6f);
#pragma unroll
  for (int i = 0; i < 4; ++i) {
    int c = t + 256 * i;
    h[(size_t)m * D_MODEL + c] = (v[i] - mu) * rstd * gamma[c] + beta[c];
  }
}

// ---------------------------------------------------------------------------
// Generic WMMA GEMM: C[M][N] = A[M][K] (lda-strided) * W[N][K]^T (+bias,+resid)
// Assumptions (all call sites satisfy): M % 128 == 0, K % 32 == 0, A fully
// in-bounds. Only N may be ragged (N=96) -> per-row guard on B staging and
// per-column guard on store.
// fp32 global -> bf16 LDS staging, f32 accumulation, double-buffered LDS,
// software-pipelined global prefetch.
// Block tile 128(M) x 128(N), BK=32. 8 waves: 2(M) x 4(N), wave = 64x32
// => 8 v_wmma per K-slab per wave.
// ---------------------------------------------------------------------------
__global__ __launch_bounds__(256) void wmma_gemm_kernel(
    const float* __restrict__ A, int lda,
    const float* __restrict__ W,
    float* __restrict__ C,
    int M, int N, int K,
    const float* __restrict__ bias,
    const float* __restrict__ resid) {
  __shared__ __align__(16) __bf16 As[2][128][40];  // row = 80B, 16B-aligned
  __shared__ __align__(16) __bf16 Bs[2][128][40];

  int t = threadIdx.x;
  int lane = t & 31;
  int wave = t >> 5;
  int wr = wave & 1;        // M direction, 2 waves
  int wc = wave >> 1;       // N direction, 4 waves
  int mWave = wr * 64;      // wave covers 64 rows
  int nWave = wc * 32;      // wave covers 32 cols
  int blockM = blockIdx.y * 128;
  int blockN = blockIdx.x * 128;

  v8f acc[4][2];
#pragma unroll
  for (int i = 0; i < 4; ++i)
#pragma unroll
    for (int j = 0; j < 2; ++j) acc[i][j] = (v8f){};

  // staging assignments: both tiles are 128 rows x 32 cols,
  // 2 threads per row, 16 contiguous floats each.
  int sRow = t >> 1;          // 0..127
  int sCol = (t & 1) * 16;    // 0 / 16

  const float* aSrc = A + (size_t)(blockM + sRow) * lda + sCol;
  const bool  bValid = (blockN + sRow) < N;
  const float* bSrc = W + (size_t)(blockN + sRow) * K + sCol;

  float4 aReg[4];
  float4 bReg[4];

  auto fetch = [&]() {
    const float4* ap = (const float4*)aSrc;
    aReg[0] = ap[0]; aReg[1] = ap[1]; aReg[2] = ap[2]; aReg[3] = ap[3];
    aSrc += 32;
    if (bValid) {
      const float4* bp = (const float4*)bSrc;
      bReg[0] = bp[0]; bReg[1] = bp[1]; bReg[2] = bp[2]; bReg[3] = bp[3];
    } else {
      bReg[0] = bReg[1] = bReg[2] = bReg[3] = make_float4(0.f, 0.f, 0.f, 0.f);
    }
    bSrc += 32;
  };

  auto pack8 = [](float4 u, float4 v) -> v8bf {
    v8bf r;
    r[0] = (__bf16)u.x; r[1] = (__bf16)u.y; r[2] = (__bf16)u.z; r[3] = (__bf16)u.w;
    r[4] = (__bf16)v.x; r[5] = (__bf16)v.y; r[6] = (__bf16)v.z; r[7] = (__bf16)v.w;
    return r;
  };

  int r15 = lane & 15;
  int hs  = lane >> 4;

  fetch();  // prologue: tile 0 -> registers

  int nk = K >> 5;
  for (int i = 0; i < nk; ++i) {
    int buf = i & 1;
    // registers -> LDS (bf16, 16B stores)
    *(v8bf*)&As[buf][sRow][sCol]     = pack8(aReg[0], aReg[1]);
    *(v8bf*)&As[buf][sRow][sCol + 8] = pack8(aReg[2], aReg[3]);
    *(v8bf*)&Bs[buf][sRow][sCol]     = pack8(bReg[0], bReg[1]);
    *(v8bf*)&Bs[buf][sRow][sCol + 8] = pack8(bReg[2], bReg[3]);
    __syncthreads();

    if (i + 1 < nk) fetch();  // overlap next tile's global loads with WMMA

    // ---- fragment loads per CDNA5 16-bit A(16x32)/B(32x16) VGPR layout ----
    v16bf afr[4], bfr[2];
    {
      int base = 8 * hs;  // lanes<16: K 0..7 / 16..23 ; lanes>=16: K 8..15 / 24..31
#pragma unroll
      for (int s = 0; s < 4; ++s) {
        const __bf16* rowp = &As[buf][mWave + 16 * s + r15][0];
#pragma unroll
        for (int q = 0; q < 8; ++q) {
          afr[s][q]     = rowp[base + q];
          afr[s][8 + q] = rowp[base + 16 + q];
        }
      }
    }
    {
      int base = 8 * hs; // lanes<16: K 0..7 / 16..23 ; lanes>=16: K 8..15 / 24..31, col = lane%16
#pragma unroll
      for (int s = 0; s < 2; ++s) {
        const __bf16* rowp = &Bs[buf][nWave + 16 * s + r15][0];
#pragma unroll
        for (int q = 0; q < 8; ++q) {
          bfr[s][q]     = rowp[base + q];
          bfr[s][8 + q] = rowp[base + 16 + q];
        }
      }
    }

#pragma unroll
    for (int s = 0; s < 4; ++s) {
      acc[s][0] = __builtin_amdgcn_wmma_f32_16x16x32_bf16(false, afr[s], false, bfr[0], (short)0, acc[s][0], false, false);
      acc[s][1] = __builtin_amdgcn_wmma_f32_16x16x32_bf16(false, afr[s], false, bfr[1], (short)0, acc[s][1], false, false);
    }
  }

  // ---- epilogue: C/D layout = 8 f32 VGPRs, lane<16: N=lane,M=r ; lane>=16: M=r+8
#pragma unroll
  for (int s = 0; s < 4; ++s) {
#pragma unroll
    for (int j = 0; j < 2; ++j) {
      int col = blockN + nWave + 16 * j + r15;
      if (col < N) {
        size_t base = (size_t)(blockM + mWave + 16 * s + 8 * hs) * N + col;
        float bv = bias ? bias[col] : 0.0f;
#pragma unroll
        for (int r = 0; r < 8; ++r) {
          float v = acc[s][j][r] + bv;
          if (resid) v += resid[base + (size_t)r * N];
          C[base + (size_t)r * N] = v;
        }
      }
    }
  }
}

// ---------------------------------------------------------------------------
// Depthwise causal conv1d (taps=4) + SiLU.
// u lives in xz[:, 0:2048] with row stride 4096. idx = m*2048 + d.
// ---------------------------------------------------------------------------
__global__ __launch_bounds__(256) void conv_silu_kernel(
    const float* __restrict__ xz,
    const float* __restrict__ conv_w,   // [D_INNER][1][4]
    const float* __restrict__ conv_b,
    float* __restrict__ uc) {
  size_t idx = (size_t)blockIdx.x * 256 + threadIdx.x;
  int d = (int)(idx & (D_INNER - 1));
  int m = (int)(idx >> 11);
  int b = m >> 11;            // L_SEQ == 2048
  int l = m & (L_SEQ - 1);
  float acc = conv_b[d];
#pragma unroll
  for (int j = 0; j < D_CONV; ++j) {
    int ls = l - (D_CONV - 1) + j;
    if (ls >= 0)
      acc += xz[((size_t)(b * L_SEQ + ls)) * (2 * D_INNER) + d] * conv_w[d * D_CONV + j];
  }
  uc[idx] = acc / (1.0f + expf(-acc));   // SiLU
}

// ---------------------------------------------------------------------------
// Selective scan: thread = (b, d) channel, 16-wide state in registers.
// delta_raw already contains dt_proj bias (GEMM epilogue); softplus here.
// ---------------------------------------------------------------------------
__global__ __launch_bounds__(256) void scan_kernel(
    const float* __restrict__ delta_raw, // [M][D_INNER]
    const float* __restrict__ xz,        // [M][2*D_INNER] (z at +D_INNER)
    const float* __restrict__ uc,        // [M][D_INNER]
    const float* __restrict__ x_dbl,     // [M][96]  (B at 64, C at 80)
    const float* __restrict__ A_log,     // [D_INNER][16]
    const float* __restrict__ Dvec,      // [D_INNER]
    float* __restrict__ y) {             // [M][D_INNER]
  int d = blockIdx.x * 256 + threadIdx.x;
  int b = blockIdx.y;
  float Av[D_STATE];
#pragma unroll
  for (int n = 0; n < D_STATE; ++n) Av[n] = -expf(A_log[d * D_STATE + n]);
  float st[D_STATE];
#pragma unroll
  for (int n = 0; n < D_STATE; ++n) st[n] = 0.0f;
  float Dd = Dvec[d];

  for (int l = 0; l < L_SEQ; ++l) {
    size_t m = (size_t)b * L_SEQ + l;
    float dlt = delta_raw[m * D_INNER + d];
    dlt = (dlt > 20.0f) ? dlt : log1pf(expf(dlt));   // softplus
    float u  = uc[m * D_INNER + d];
    float du = dlt * u;
    const float* bc = x_dbl + m * XDBL_N;
    float accy = 0.0f;
#pragma unroll
    for (int n = 0; n < D_STATE; ++n) {
      float dA = expf(dlt * Av[n]);
      st[n] = dA * st[n] + du * bc[DT_RANK + n];
      accy += st[n] * bc[DT_RANK + D_STATE + n];
    }
    float yv = accy + u * Dd;
    float z = xz[m * (2 * D_INNER) + D_INNER + d];
    yv *= z / (1.0f + expf(-z));                     // * silu(z)
    y[m * D_INNER + d] = yv;
  }
}

// ---------------------------------------------------------------------------
extern "C" void kernel_launch(void* const* d_in, const int* in_sizes, int n_in,
                              void* d_out, int out_size, void* d_ws, size_t ws_size,
                              hipStream_t stream) {
  const float* x         = (const float*)d_in[0];
  const float* ln_gamma  = (const float*)d_in[1];
  const float* ln_beta   = (const float*)d_in[2];
  const float* in_proj_w = (const float*)d_in[3];
  const float* conv_w    = (const float*)d_in[4];
  const float* conv_b    = (const float*)d_in[5];
  const float* x_proj_w  = (const float*)d_in[6];
  const float* dt_proj_w = (const float*)d_in[7];
  const float* dt_proj_b = (const float*)d_in[8];
  const float* A_log     = (const float*)d_in[9];
  const float* Dvec      = (const float*)d_in[10];
  const float* out_proj_w= (const float*)d_in[11];
  float* out = (float*)d_out;

  float* ws = (float*)d_ws;
  size_t off = 0;
  float* h     = ws + off; off += (size_t)M_ROWS * D_MODEL;      // 4M
  float* xz    = ws + off; off += (size_t)M_ROWS * 2 * D_INNER;  // 16M
  float* uc    = ws + off; off += (size_t)M_ROWS * D_INNER;      // 8M
  float* x_dbl = ws + off; off += (size_t)M_ROWS * XDBL_N;       // 0.4M
  float* delta = ws + off; off += (size_t)M_ROWS * D_INNER;      // 8M
  float* ybuf  = ws + off; off += (size_t)M_ROWS * D_INNER;      // 8M
  (void)ws_size; (void)in_sizes; (void)n_in; (void)out_size;

  // 1) LayerNorm
  ln_kernel<<<M_ROWS, 256, 0, stream>>>(x, ln_gamma, ln_beta, h);

  // 2) in_proj: xz[4096][4096] = h[4096][1024] * in_proj_w[4096][1024]^T
  wmma_gemm_kernel<<<dim3((2 * D_INNER) / 128, M_ROWS / 128), 256, 0, stream>>>(
      h, D_MODEL, in_proj_w, xz, M_ROWS, 2 * D_INNER, D_MODEL, nullptr, nullptr);

  // 3) causal depthwise conv + SiLU
  conv_silu_kernel<<<(M_ROWS * D_INNER) / 256, 256, 0, stream>>>(xz, conv_w, conv_b, uc);

  // 4) x_proj: x_dbl[4096][96] = uc * x_proj_w[96][2048]^T
  wmma_gemm_kernel<<<dim3((XDBL_N + 127) / 128, M_ROWS / 128), 256, 0, stream>>>(
      uc, D_INNER, x_proj_w, x_dbl, M_ROWS, XDBL_N, D_INNER, nullptr, nullptr);

  // 5) dt_proj: delta[4096][2048] = x_dbl[:, :64] * dt_proj_w[2048][64]^T + dt_proj_b
  wmma_gemm_kernel<<<dim3(D_INNER / 128, M_ROWS / 128), 256, 0, stream>>>(
      x_dbl, XDBL_N, dt_proj_w, delta, M_ROWS, D_INNER, DT_RANK, dt_proj_b, nullptr);

  // 6) selective scan + skip + gating
  scan_kernel<<<dim3(D_INNER / 256, B_SZ), 256, 0, stream>>>(
      delta, xz, uc, x_dbl, A_log, Dvec, ybuf);

  // 7) out_proj + residual: out[4096][1024] = ybuf * out_proj_w[1024][2048]^T + x
  wmma_gemm_kernel<<<dim3(D_MODEL / 128, M_ROWS / 128), 256, 0, stream>>>(
      ybuf, D_INNER, out_proj_w, out, M_ROWS, D_MODEL, D_INNER, nullptr, x);
}
